// RelationNetwork_53584011985126
// MI455X (gfx1250) — matrix operand order repaired and checked by reading, not compiled, not run on hardware
//
#include <hip/hip_runtime.h>
#include <hip/hip_bf16.h>

// Relation Network for MI455X (gfx1250, wave32, WMMA).
// B=32, C=256, N=64 objects, HG=HF=512, D_OUT=256.
//
// Layer0 factorized into per-object projections (~1 GFLOP VALU);
// layers 1&2 = 137 GFLOP of bf16 WMMA (hot path); f-MLP negligible.
// v3: fix async builtin signature: (int4 AS1* gsrc, int4 AS3* ldst, imm, imm).

#define BB   32
#define NOBJ 64
#define CC   256
#define HGD  512

#define HP   520                 // padded h-row stride (bf16): 260 dwords == 4 mod 64
#define WP   40                  // padded weight-chunk row (bf16): 20 dwords, conflict-free
#define WCHUNK (HGD * WP)        // one staged K-chunk: 512 rows x 40 bf16 = 40KB
#define LDS_BYTES (2 * NOBJ * HP * 2 + 2 * WCHUNK * 2)   // 133120 + 81920 = 215040

typedef __attribute__((ext_vector_type(16))) __bf16 v16bf;
typedef __attribute__((ext_vector_type(8)))  float  v8f;
typedef int v4i_ __attribute__((vector_size(16)));      // matches builtin param type

union AB32 { v16bf v; uint4 q[2]; };   // 16 bf16 = 32B A/B fragment per lane

static __device__ inline __bf16 to_bf16(float f) { return (__bf16)f; }

#if defined(__AMDGCN__) && __has_builtin(__builtin_amdgcn_global_load_async_to_lds_b128)
#define ASYNC_STAGE 1
#else
#define ASYNC_STAGE 0
#endif

static __device__ inline void async_fence()
{
#if ASYNC_STAGE
#if __has_builtin(__builtin_amdgcn_s_wait_asynccnt)
    __builtin_amdgcn_s_wait_asynccnt(0);
#else
    asm volatile("s_wait_asynccnt 0" ::: "memory");
#endif
#endif
}

// ---------------------------------------------------------------------------
// Prep: transpose+convert g_W1, g_W2 (f32 [K][N]) -> bf16 [N][K] for B-frag
// streaming (each lane's WMMA B fragment is then two 16B runs of one row).
// ---------------------------------------------------------------------------
__global__ __launch_bounds__(256)
void prep_weights(const float* __restrict__ gW1, const float* __restrict__ gW2,
                  __bf16* __restrict__ W1t, __bf16* __restrict__ W2t)
{
    int idx = blockIdx.x * 256 + threadIdx.x;     // 0 .. 512*512-1
    int n = idx >> 9, k = idx & (HGD - 1);        // coalesced writes
    W1t[n * HGD + k] = to_bf16(gW1[k * HGD + n]);
    W2t[n * HGD + k] = to_bf16(gW2[k * HGD + n]);
}

// ---------------------------------------------------------------------------
// Layer-0 factorization: Abuf[b][j][h] = sum_c x[b][c][j] * W0[c][h]
//                        Bmbuf[b][j][h] = sum_c x[b][c][j] * W0[256+c][h]
// Only 2048 rows instead of 131072 -> saves 68 GFLOP.
// ---------------------------------------------------------------------------
__global__ __launch_bounds__(256)
void compute_ab(const float* __restrict__ x, const float* __restrict__ gW0,
                float* __restrict__ Abuf, float* __restrict__ Bmbuf)
{
    __shared__ float xs[CC];
    int b = blockIdx.x >> 6, j = blockIdx.x & 63;
    int t = threadIdx.x;
    xs[t] = x[(b * CC + t) * NOBJ + j];           // x[b][c=t][j]
    __syncthreads();
    #pragma unroll
    for (int hh = 0; hh < 2; ++hh) {
        int h = t + hh * 256;
        float a = 0.f, bm = 0.f;
        for (int c = 0; c < CC; ++c) {
            float xv = xs[c];
            a  = fmaf(xv, gW0[c * HGD + h], a);            // W0 rows 0..255 (x_q side)
            bm = fmaf(xv, gW0[(CC + c) * HGD + h], bm);    // W0 rows 256..511 (x_p side)
        }
        Abuf [(b * NOBJ + j) * HGD + h] = a;
        Bmbuf[(b * NOBJ + j) * HGD + h] = bm;
    }
}

// ---------------------------------------------------------------------------
// Main fused kernel: one workgroup per (b, p-pair). 512 threads = 16 waves.
// Rows = 128 pairs: (p0, q=0..63) then (p1, q=0..63).
//   1) build h0 (128 x 512 bf16, padded rows) in LDS
//   2) h1 = ReLU(h0 @ W1 + b1)  -- WMMA, W1t double-buffered in LDS
//   3) rel = ReLU(h1 @ W2 + b2) -- WMMA, rows reduced in-register
//   4) partial[b][p][h] = sum_q rel
// Wave w owns a 64(M) x 64(N) block: mhalf = w>>3 picks the p row-half,
// (w&7) picks 4 of 32 N-tiles -> 4x4 v8f accumulators (128 VGPRs).
// h1 overwrites h0's buffer (dead after layer-1 K loop).
// ---------------------------------------------------------------------------
__global__ __launch_bounds__(512, 1)
void relnet_main(const float* __restrict__ Abuf, const float* __restrict__ Bmbuf,
                 const float* __restrict__ gW0, const float* __restrict__ gb0,
                 const __bf16* __restrict__ W1t, const float* __restrict__ gb1,
                 const __bf16* __restrict__ W2t, const float* __restrict__ gb2,
                 float* __restrict__ partial)
{
    extern __shared__ char smem[];
    __bf16* hbuf = (__bf16*)smem;                        // 128 x HP bf16 (h0, then h1)
    __bf16* wbuf = (__bf16*)(smem + 2 * NOBJ * HP * 2);  // 2 x WCHUNK bf16 weight stage

    const int b     = blockIdx.x >> 5;
    const int p0    = (blockIdx.x & 31) * 2;
    const int t     = threadIdx.x;          // 0..511
    const int wave  = t >> 5;               // 0..15
    const int lane  = t & 31;
    const int lm    = lane & 15;            // M (A/C row) or N (B/C col) within tile
    const int lh    = lane >> 4;            // lane-half selects K/M sub-range
    const int mhalf = wave >> 3;            // which p this wave reduces
    const int nwav  = wave & 7;             // N-tile group: ntiles nwav*4 .. +3

    // ---- 1) build h0 rows: ReLU(A[q] + Bm[p] + rel(p,q)@W0c + b0) -> bf16
    // rel.mean() == 0 exactly (antisymmetric), so rel = raw grid deltas.
    #pragma unroll 4
    for (int i = 0; i < (2 * NOBJ * HGD) / 512; ++i) {
        int idx = t + i * 512;
        int row = idx >> 9;                  // 0..127
        int h   = idx & (HGD - 1);
        int q   = row & 63;
        int pv  = p0 + (row >> 6);
        float r0 = (float)(pv >> 3) - (float)(q >> 3);
        float r1 = (float)(pv & 7)  - (float)(q & 7);
        float v = Abuf [(b * NOBJ + q ) * HGD + h]
                + Bmbuf[(b * NOBJ + pv) * HGD + h]
                + r0 * gW0[512 * HGD + h]
                + r1 * gW0[513 * HGD + h]
                + gb0[h];
        hbuf[row * HP + h] = to_bf16(fmaxf(v, 0.f));
    }
    __syncthreads();

    v8f acc[4][4];

    for (int layer = 0; layer < 2; ++layer) {
        const __bf16* Wt   = layer ? W2t : W1t;
        const float*  bias = layer ? gb2 : gb1;

        // stage Wt[n][kk..kk+31] -> wbuf[buf], layout [n][WP] (B-frag friendly)
        auto stage = [&](int buf, int kk) {
            const __bf16* src = Wt + kk;
            __bf16* dst = wbuf + buf * WCHUNK;
            #pragma unroll
            for (int i = 0; i < 4; ++i) {
                int idx = t + i * 512;       // 2048 x 16B chunks = 32KB payload
                int n = idx >> 2, seg = idx & 3;
#if ASYNC_STAGE
                __builtin_amdgcn_global_load_async_to_lds_b128(
                    (__attribute__((address_space(1))) v4i_*)(src + n * HGD + seg * 8),
                    (__attribute__((address_space(3))) v4i_*)(dst + n * WP + seg * 8),
                    0, 0);
#else
                *(uint4*)(dst + n * WP + seg * 8) =
                    *(const uint4*)(src + n * HGD + seg * 8);
#endif
            }
        };

        #pragma unroll
        for (int m = 0; m < 4; ++m)
            #pragma unroll
            for (int n = 0; n < 4; ++n) {
                v8f z = {0.f,0.f,0.f,0.f,0.f,0.f,0.f,0.f};
                acc[m][n] = z;
            }

        stage(0, 0);
        async_fence();
        __syncthreads();

        for (int ks = 0; ks < HGD / 32; ++ks) {
            int cur = ks & 1;
            if (ks + 1 < HGD / 32) stage(cur ^ 1, (ks + 1) * 32);

            // A fragments: 16x32 bf16, row-major (padded) in LDS.
            // ISA layout: lanes 0-15 hold K {0..7,16..23}, lanes 16-31 {8..15,24..31}.
            const int k0 = ks * 32;
            AB32 aF[4];
            #pragma unroll
            for (int m = 0; m < 4; ++m) {
                const __bf16* r = hbuf + (mhalf * 64 + m * 16 + lm) * HP + k0;
                aF[m].q[0] = *(const uint4*)(r + (lh ? 8  : 0));
                aF[m].q[1] = *(const uint4*)(r + (lh ? 24 : 16));
            }

            // B fragments: 32x16 bf16; lanes 0-15 hold K {0..7,16..23}, lanes 16-31 {8..15,24..31}
            // -> two 16B reads per lane from the [n][WP] staged chunk.
            const __bf16* wb = wbuf + cur * WCHUNK;
            #pragma unroll
            for (int n = 0; n < 4; ++n) {
                int col = (nwav * 4 + n) * 16 + lm;
                AB32 bF;
                const __bf16* r = wb + col * WP + (lh ? 8 : 0);
                bF.q[0] = *(const uint4*)(r);
                bF.q[1] = *(const uint4*)(r + 16);
                #pragma unroll
                for (int m = 0; m < 4; ++m)
                    acc[m][n] = __builtin_amdgcn_wmma_f32_16x16x32_bf16(
                        false, aF[m].v, false, bF.v, (short)0, acc[m][n],
                        false, false);
            }
            async_fence();
            __syncthreads();   // double-buffer handoff
        }

        if (layer == 0) {
            // epilogue: bias + ReLU -> bf16, write h1 over h0's buffer.
            // C layout: lane l, vgpr r -> row = 8*(l>=16)+r, col = l&15.
            #pragma unroll
            for (int n = 0; n < 4; ++n) {
                int col = (nwav * 4 + n) * 16 + lm;
                float bi = bias[col];
                #pragma unroll
                for (int m = 0; m < 4; ++m)
                    #pragma unroll
                    for (int r = 0; r < 8; ++r) {
                        int row = mhalf * 64 + m * 16 + lh * 8 + r;
                        hbuf[row * HP + col] =
                            to_bf16(fmaxf(acc[m][n][r] + bi, 0.f));
                    }
            }
            __syncthreads();
        } else {
            // epilogue: bias + ReLU + sum over this p-half's 64 rows (q)
            #pragma unroll
            for (int n = 0; n < 4; ++n) {
                int col = (nwav * 4 + n) * 16 + lm;
                float bi = bias[col];
                float s = 0.f;
                #pragma unroll
                for (int m = 0; m < 4; ++m)
                    #pragma unroll
                    for (int r = 0; r < 8; ++r)
                        s += fmaxf(acc[m][n][r] + bi, 0.f);
                s += __shfl_xor(s, 16, 32);        // wave32: fold row halves
                if (lh == 0)
                    partial[(size_t)(b * NOBJ + p0 + mhalf) * HGD + col] = s;
            }
        }
    }
}

// ---------------------------------------------------------------------------
// Reduce over p: emb[b][h] = sum_p partial[b][p][h]
// ---------------------------------------------------------------------------
__global__ __launch_bounds__(256)
void reduce_pairs(const float* __restrict__ partial, float* __restrict__ emb)
{
    int idx = blockIdx.x * 256 + threadIdx.x;     // 32*512
    int b = idx >> 9, h = idx & (HGD - 1);
    float s = 0.f;
    for (int pp = 0; pp < NOBJ; ++pp)
        s += partial[(b * NOBJ + pp) * HGD + h];
    emb[idx] = s;
}

// ---------------------------------------------------------------------------
// f MLP (tiny: ~25 MFLOP total) — plain VALU.
// ---------------------------------------------------------------------------
__global__ __launch_bounds__(256)
void f_mlp1(const float* __restrict__ emb, const float* __restrict__ fW0,
            const float* __restrict__ fb0, float* __restrict__ hf)
{
    int idx = blockIdx.x * 256 + threadIdx.x;     // 32*512
    int b = idx >> 9, j = idx & (HGD - 1);
    float s = fb0[j];
    const float* e = emb + b * HGD;
    for (int h = 0; h < HGD; ++h) s = fmaf(e[h], fW0[h * HGD + j], s);
    hf[idx] = fmaxf(s, 0.f);
}

__global__ __launch_bounds__(256)
void f_mlp2(const float* __restrict__ hf, const float* __restrict__ fW1,
            const float* __restrict__ fb1, float* __restrict__ out)
{
    int idx = blockIdx.x * 256 + threadIdx.x;     // 32*256
    int b = idx >> 8, j = idx & 255;
    float s = fb1[j];
    const float* e = hf + b * HGD;
    for (int h = 0; h < HGD; ++h) s = fmaf(e[h], fW1[h * 256 + j], s);
    out[idx] = s;
}

// ---------------------------------------------------------------------------
extern "C" void kernel_launch(void* const* d_in, const int* in_sizes, int n_in,
                              void* d_out, int out_size, void* d_ws, size_t ws_size,
                              hipStream_t stream)
{
    (void)in_sizes; (void)n_in; (void)out_size; (void)ws_size;

    const float* x   = (const float*)d_in[0];
    const float* gW0 = (const float*)d_in[1];
    const float* gb0 = (const float*)d_in[2];
    const float* gW1 = (const float*)d_in[3];
    const float* gb1 = (const float*)d_in[4];
    const float* gW2 = (const float*)d_in[5];
    const float* gb2 = (const float*)d_in[6];
    const float* fW0 = (const float*)d_in[7];
    const float* fb0 = (const float*)d_in[8];
    const float* fW1 = (const float*)d_in[9];
    const float* fb1 = (const float*)d_in[10];

    // workspace carve-up (~13.8 MB)
    char* ws = (char*)d_ws;
    __bf16* W1t    = (__bf16*)ws;  ws += (size_t)HGD * HGD * 2;
    __bf16* W2t    = (__bf16*)ws;  ws += (size_t)HGD * HGD * 2;
    float* Abuf    = (float*)ws;   ws += (size_t)BB * NOBJ * HGD * 4;
    float* Bmbuf   = (float*)ws;   ws += (size_t)BB * NOBJ * HGD * 4;
    float* partial = (float*)ws;   ws += (size_t)BB * NOBJ * HGD * 4;
    float* emb     = (float*)ws;   ws += (size_t)BB * HGD * 4;
    float* hf      = (float*)ws;   ws += (size_t)BB * HGD * 4;

    prep_weights<<<dim3((HGD * HGD) / 256), dim3(256), 0, stream>>>(gW1, gW2, W1t, W2t);
    compute_ab  <<<dim3(BB * NOBJ), dim3(256), 0, stream>>>(x, gW0, Abuf, Bmbuf);
    relnet_main <<<dim3(BB * (NOBJ / 2)), dim3(512), LDS_BYTES, stream>>>(
        Abuf, Bmbuf, gW0, gb0, W1t, gb1, W2t, gb2, partial);
    reduce_pairs<<<dim3((BB * HGD) / 256), dim3(256), 0, stream>>>(partial, emb);
    f_mlp1      <<<dim3((BB * HGD) / 256), dim3(256), 0, stream>>>(emb, fW0, fb0, hf);
    f_mlp2      <<<dim3((BB * 256) / 256), dim3(256), 0, stream>>>(hf, fW1, fb1, (float*)d_out);
}
